// LlamaModeling_73349451481402
// MI455X (gfx1250) — matrix operand order repaired and checked by reading, not compiled, not run on hardware
//
#include <hip/hip_runtime.h>

// ---------------------------------------------------------------------------
// MI455X (gfx1250, wave32) implementation of a fused Llama block.
// All GEMM-shaped math runs through v_wmma_f32_16x16x32_bf16.
// Global->LDS tile movement uses gfx1250 async-to-LDS ops (ASYNCcnt-tracked)
// via inline asm, plus global_prefetch for look-ahead.
// ---------------------------------------------------------------------------

typedef __bf16 bf16;
typedef __attribute__((ext_vector_type(16))) __bf16 v16bf;
typedef __attribute__((ext_vector_type(8)))  float  v8f;

#define NHEAD 16
#define HDIM  128
#define DMODEL 2048
#define FFDIM  8192
#define SEQ    2048
#define BATCH  2

// ---- gfx1250 async global->LDS ---------------------------------------------
// The builtins exist on this toolchain (__has_builtin confirmed) but take
// typed address-space pointers; use inline asm to emit the instruction
// directly: VDST = VGPR holding LDS byte address, VADDR = 64-bit global addr.
#if __has_builtin(__builtin_amdgcn_global_load_async_to_lds_b128) && \
    __has_builtin(__builtin_amdgcn_s_wait_asynccnt)
#define USE_ASYNC 1
__device__ __forceinline__ void async_cp16(const void* g, void* l) {
    // low 32 bits of a generic shared pointer are the LDS byte address
    unsigned ldsoff = (unsigned)(size_t)l;
    asm volatile("global_load_async_to_lds_b128 %0, %1, off"
                 :: "v"(ldsoff), "v"((unsigned long long)(size_t)g)
                 : "memory");
}
__device__ __forceinline__ void async_fence() { __builtin_amdgcn_s_wait_asynccnt(0); }
#else
#define USE_ASYNC 0
__device__ __forceinline__ void async_fence() {}
#endif

__device__ __forceinline__ bf16 f2bf(float f) {
    union { float f; unsigned int u; } a; a.f = f;
    unsigned int r = a.u + 0x7FFFu + ((a.u >> 16) & 1u);
    union { unsigned short s; bf16 b; } c; c.s = (unsigned short)(r >> 16);
    return c.b;
}

// ---------------------------------------------------------------------------
// fp32 -> bf16 conversion (weights / adapter)
// ---------------------------------------------------------------------------
__global__ void cvt_bf16_kernel(const float* __restrict__ in, bf16* __restrict__ out, long n) {
    long i = (long)blockIdx.x * blockDim.x + threadIdx.x;
    if (i < n) out[i] = f2bf(in[i]);
}

// ---------------------------------------------------------------------------
// RMSNorm: one 256-thread block per row of d=2048, bf16 output
// ---------------------------------------------------------------------------
__global__ __launch_bounds__(256) void rmsnorm_kernel(
    const float* __restrict__ x, const float* __restrict__ w,
    bf16* __restrict__ out, int d)
{
    __shared__ float red[8];
    const float* row = x + (long)blockIdx.x * d;
    float s = 0.f;
    for (int i = threadIdx.x; i < d; i += 256) { float v = row[i]; s += v * v; }
    #pragma unroll
    for (int off = 16; off > 0; off >>= 1) s += __shfl_xor(s, off, 32);
    if ((threadIdx.x & 31) == 0) red[threadIdx.x >> 5] = s;
    __syncthreads();
    if (threadIdx.x == 0) { float t = 0.f; for (int i = 0; i < 8; i++) t += red[i]; red[0] = t; }
    __syncthreads();
    float inv = rsqrtf(red[0] / (float)d + 1e-6f);
    bf16* orow = out + (long)blockIdx.x * d;
    for (int i = threadIdx.x; i < d; i += 256) orow[i] = f2bf(row[i] * inv * w[i]);
}

// ---------------------------------------------------------------------------
// Generic bf16 WMMA GEMM:  C[M,N] = A[M,K] (row-major) * B[N,K]^T (row-major)
// Block tile 128x128, BK=32, 8 waves (4x2 over MxN), wave tile 32x64.
// Epilogues: 0 = store f32, 1 = store f32 + aux (residual),
//            2 = store bf16 of silu(aux)*acc (SwiGLU fusion)
// ---------------------------------------------------------------------------
#define BM 128
#define BN 128
#define BK 32

__global__ __launch_bounds__(256) void gemm_bf16_wmma(
    const bf16* __restrict__ A, const bf16* __restrict__ B,
    int M, int N, int K, int mode,
    float* __restrict__ outf, bf16* __restrict__ outbf,
    const float* __restrict__ aux)
{
    __shared__ __align__(16) bf16 As[2][BM * BK];
    __shared__ __align__(16) bf16 Bs[2][BN * BK];

    const int tid  = threadIdx.x;
    const int lane = tid & 31;
    const int hf   = lane >> 4;   // lane half (0/1)
    const int l16  = lane & 15;
    const int wave = tid >> 5;
    const int waveM = wave >> 1;  // 0..3
    const int waveN = wave & 1;   // 0..1
    const int bm = blockIdx.y * BM;
    const int bn = blockIdx.x * BN;

    v8f acc[2][4];
    #pragma unroll
    for (int i = 0; i < 2; i++)
        #pragma unroll
        for (int j = 0; j < 4; j++) acc[i][j] = {};

    const int nk = K / BK;
    const int ra = tid >> 2, ca = (tid & 3) * 8;

    auto rowA0 = [&](void) { int g = bm + ra;      return (g >= M) ? M - 1 : g; };
    auto rowA1 = [&](void) { int g = bm + ra + 64; return (g >= M) ? M - 1 : g; };

    auto prefetchTile = [&](int kt) {
        int k0 = kt * BK;
        __builtin_prefetch(A + (long)rowA0() * K + k0 + ca, 0, 1);
        __builtin_prefetch(B + (long)(bn + ra) * K + k0 + ca, 0, 1);
    };

#if USE_ASYNC
    auto asyncTile = [&](int buf, int kt) {
        int k0 = kt * BK;
        async_cp16(A + (long)rowA0() * K + k0 + ca, &As[buf][ra * BK + ca]);
        async_cp16(A + (long)rowA1() * K + k0 + ca, &As[buf][(ra + 64) * BK + ca]);
        async_cp16(B + (long)(bn + ra) * K + k0 + ca,      &Bs[buf][ra * BK + ca]);
        async_cp16(B + (long)(bn + ra + 64) * K + k0 + ca, &Bs[buf][(ra + 64) * BK + ca]);
    };
#else
    auto loadAB = [&](int kt, uint4& a0, uint4& a1, uint4& b0, uint4& b1) {
        int k0 = kt * BK;
        a0 = *(const uint4*)(A + (long)rowA0() * K + k0 + ca);
        a1 = *(const uint4*)(A + (long)rowA1() * K + k0 + ca);
        b0 = *(const uint4*)(B + (long)(bn + ra) * K + k0 + ca);
        b1 = *(const uint4*)(B + (long)(bn + ra + 64) * K + k0 + ca);
    };
    auto storeLds = [&](int buf, uint4 a0, uint4 a1, uint4 b0, uint4 b1) {
        *(uint4*)&As[buf][ra * BK + ca]        = a0;
        *(uint4*)&As[buf][(ra + 64) * BK + ca] = a1;
        *(uint4*)&Bs[buf][ra * BK + ca]        = b0;
        *(uint4*)&Bs[buf][(ra + 64) * BK + ca] = b1;
    };
#endif

    union FragU { v16bf v; uint4 q[2]; };

    auto compute = [&](int buf) {
        FragU fa[2], fb[4];
        #pragma unroll
        for (int mt = 0; mt < 2; mt++) {
            const bf16* p = &As[buf][(waveM * 32 + mt * 16 + l16) * BK + hf * 8];
            fa[mt].q[0] = *(const uint4*)p;
            fa[mt].q[1] = *(const uint4*)(p + 16);
        }
        #pragma unroll
        for (int nt = 0; nt < 4; nt++) {
            const bf16* p = &Bs[buf][(waveN * 64 + nt * 16 + l16) * BK + hf * 8];
            fb[nt].q[0] = *(const uint4*)p;
            fb[nt].q[1] = *(const uint4*)(p + 16);
        }
        #pragma unroll
        for (int mt = 0; mt < 2; mt++)
            #pragma unroll
            for (int nt = 0; nt < 4; nt++)
                acc[mt][nt] = __builtin_amdgcn_wmma_f32_16x16x32_bf16(
                    false, fa[mt].v, false, fb[nt].v, (short)0, acc[mt][nt], false, false);
    };

#if USE_ASYNC
    asyncTile(0, 0);
    async_fence();
    __syncthreads();
    for (int kt = 0; kt < nk; kt++) {
        int cur = kt & 1;
        bool pre = (kt + 1 < nk);
        if (pre) asyncTile(cur ^ 1, kt + 1);
        if (kt + 2 < nk) prefetchTile(kt + 2);
        compute(cur);
        if (pre) { async_fence(); __syncthreads(); }
    }
#else
    { // prologue
        uint4 a0, a1, b0, b1;
        loadAB(0, a0, a1, b0, b1);
        storeLds(0, a0, a1, b0, b1);
    }
    __syncthreads();
    for (int kt = 0; kt < nk; kt++) {
        int cur = kt & 1;
        uint4 a0, a1, b0, b1;
        bool pre = (kt + 1 < nk);
        if (pre) loadAB(kt + 1, a0, a1, b0, b1);
        if (kt + 2 < nk) prefetchTile(kt + 2);
        compute(cur);
        if (pre) {
            __syncthreads();
            storeLds(cur ^ 1, a0, a1, b0, b1);
            __syncthreads();
        }
    }
#endif

    // epilogue
    #pragma unroll
    for (int mt = 0; mt < 2; mt++) {
        #pragma unroll
        for (int nt = 0; nt < 4; nt++) {
            int col = bn + waveN * 64 + nt * 16 + l16;
            #pragma unroll
            for (int j = 0; j < 8; j++) {
                int row = bm + waveM * 32 + mt * 16 + j + 8 * hf;
                if (row < M) {
                    long idx = (long)row * N + col;
                    float v = acc[mt][nt][j];
                    if (mode == 0)      outf[idx] = v;
                    else if (mode == 1) outf[idx] = v + aux[idx];
                    else {
                        float g = aux[idx];
                        float s = g / (1.f + __expf(-g));
                        outbf[idx] = f2bf(s * v);
                    }
                }
            }
        }
    }
}

// ---------------------------------------------------------------------------
// RoPE on q/k (f32 [b,s,h*d]) -> bf16 [b,h,s,d]; fold 1/sqrt(HD) into q
// ---------------------------------------------------------------------------
__global__ __launch_bounds__(64) void rope_kernel(
    const float* __restrict__ q, const float* __restrict__ k,
    bf16* __restrict__ qo, bf16* __restrict__ ko, float scale)
{
    int idx = blockIdx.x;
    int s = idx % SEQ;
    int h = (idx / SEQ) % NHEAD;
    int b = idx / (SEQ * NHEAD);
    int i = threadIdx.x; // 0..63
    long inoff = ((long)(b * SEQ + s)) * DMODEL + h * HDIM;
    float inv = __powf(10000.0f, -(float)i / 64.0f);
    float ang = (float)s * inv;
    float c = __cosf(ang), sn = __sinf(ang);
    float q1 = q[inoff + i], q2 = q[inoff + i + 64];
    float k1 = k[inoff + i], k2 = k[inoff + i + 64];
    long ooff = (((long)b * NHEAD + h) * SEQ + s) * HDIM;
    qo[ooff + i]      = f2bf((q1 * c - q2 * sn) * scale);
    qo[ooff + i + 64] = f2bf((q2 * c + q1 * sn) * scale);
    ko[ooff + i]      = f2bf(k1 * c - k2 * sn);
    ko[ooff + i + 64] = f2bf(k2 * c + k1 * sn);
}

// V repack: f32 [b,s,h,d] -> bf16 [b,h,s,d]
__global__ __launch_bounds__(128) void vpack_kernel(const float* __restrict__ v, bf16* __restrict__ vo)
{
    int idx = blockIdx.x;
    int s = idx % SEQ;
    int h = (idx / SEQ) % NHEAD;
    int b = idx / (SEQ * NHEAD);
    int d = threadIdx.x;
    vo[(((long)b * NHEAD + h) * SEQ + s) * HDIM + d] =
        f2bf(v[((long)(b * SEQ + s)) * DMODEL + h * HDIM + d]);
}

// adapter proj pack: f32 [b*10, 2048] -> bf16 [b,h,32,d] (rows >= 10 zero)
__global__ __launch_bounds__(128) void adpack_kernel(const float* __restrict__ p, bf16* __restrict__ o)
{
    int idx = blockIdx.x;
    int t = idx % 32;
    int h = (idx / 32) % NHEAD;
    int b = idx / (32 * NHEAD);
    int d = threadIdx.x;
    float val = (t < 10) ? p[((long)(b * 10 + t)) * DMODEL + h * HDIM + d] : 0.f;
    o[((((long)b * NHEAD + h) * 32) + t) * HDIM + d] = f2bf(val);
}

// ---------------------------------------------------------------------------
// Flash attention (causal) + gated adapter attention.
// 128 threads = 4 waves; each wave owns 16 query rows. 32-key tiles.
// Output written as bf16 [b, s, h*d] for the following O-projection GEMM.
// ---------------------------------------------------------------------------
#define FA_KT 32

__global__ __launch_bounds__(128) void flash_attn_kernel(
    const bf16* __restrict__ Q, const bf16* __restrict__ Kmat,
    const bf16* __restrict__ V, const bf16* __restrict__ AK,
    const bf16* __restrict__ AV, const float* __restrict__ gate,
    bf16* __restrict__ Out)
{
    __shared__ __align__(16) bf16 Ks[2][FA_KT * HDIM];  // [key][d]
    __shared__ __align__(16) bf16 Vs[2][HDIM * FA_KT];  // [d][key] (transposed)
    __shared__ __align__(16) bf16 Ps[4][16 * 32];       // per-wave P scratch

    const int tid = threadIdx.x, lane = tid & 31, wave = tid >> 5;
    const int hf = lane >> 4, l16 = lane & 15;
    const int nqt = SEQ / 64;
    const int qt = blockIdx.x % nqt;
    const int h  = (blockIdx.x / nqt) % NHEAD;
    const int b  = blockIdx.x / (nqt * NHEAD);
    const long bh = (long)b * NHEAD + h;
    const bf16* Qp = Q    + bh * SEQ * HDIM;
    const bf16* Kp = Kmat + bh * SEQ * HDIM;
    const bf16* Vp = V    + bh * SEQ * HDIM;
    const int q0 = qt * 64;
    const int qrow = q0 + wave * 16;

    union FragU { v16bf v; uint4 q[2]; };

    // Q fragments (scale already folded in)
    FragU fq[4];
    #pragma unroll
    for (int kc = 0; kc < 4; kc++) {
        const bf16* p = Qp + (long)(qrow + l16) * HDIM + kc * 32 + hf * 8;
        fq[kc].q[0] = *(const uint4*)p;
        fq[kc].q[1] = *(const uint4*)(p + 16);
    }

    v8f o[8];
    #pragma unroll
    for (int i = 0; i < 8; i++) o[i] = {};
    float m[8], l[8];
    #pragma unroll
    for (int j = 0; j < 8; j++) { m[j] = -3.0e38f; l[j] = 0.f; }

    auto loadKV = [&](int buf, int kbase, const bf16* kp, const bf16* vp) {
        #pragma unroll
        for (int i = 0; i < 4; i++) {
            int c = tid + 128 * i;
            int r = c >> 4, col = (c & 15) * 8;
#if USE_ASYNC
            async_cp16(kp + (long)(kbase + r) * HDIM + col, &Ks[buf][r * HDIM + col]);
#else
            *(uint4*)&Ks[buf][r * HDIM + col] = *(const uint4*)(kp + (long)(kbase + r) * HDIM + col);
#endif
            union { uint4 u; bf16 e[8]; } t;
            t.u = *(const uint4*)(vp + (long)(kbase + r) * HDIM + col);
            #pragma unroll
            for (int e = 0; e < 8; e++) Vs[buf][(col + e) * FA_KT + r] = t.e[e];
        }
    };

    auto scoreTiles = [&](int buf, v8f* s) {
        #pragma unroll
        for (int nt = 0; nt < 2; nt++) {
            s[nt] = {};
            #pragma unroll
            for (int kc = 0; kc < 4; kc++) {
                FragU fk;
                const bf16* p = &Ks[buf][(nt * 16 + l16) * HDIM + kc * 32 + hf * 8];
                fk.q[0] = *(const uint4*)p;
                fk.q[1] = *(const uint4*)(p + 16);
                s[nt] = __builtin_amdgcn_wmma_f32_16x16x32_bf16(
                    false, fq[kc].v, false, fk.v, (short)0, s[nt], false, false);
            }
        }
    };

    auto pvAccum = [&](int buf) {
        bf16* Pw = &Ps[wave][0];
        asm volatile("s_wait_dscnt 0" ::: "memory");
        FragU fp;
        const bf16* pp = Pw + l16 * 32 + hf * 8;
        fp.q[0] = *(const uint4*)pp;
        fp.q[1] = *(const uint4*)(pp + 16);
        #pragma unroll
        for (int dt = 0; dt < 8; dt++) {
            FragU fv;
            const bf16* vp = &Vs[buf][(dt * 16 + l16) * FA_KT + hf * 8];
            fv.q[0] = *(const uint4*)vp;
            fv.q[1] = *(const uint4*)(vp + 16);
            o[dt] = __builtin_amdgcn_wmma_f32_16x16x32_bf16(
                false, fp.v, false, fv.v, (short)0, o[dt], false, false);
        }
    };

    const int nkt = (q0 + 64) / FA_KT;
    loadKV(0, 0, Kp, Vp);
    async_fence();
    __syncthreads();

    for (int kt = 0; kt < nkt; kt++) {
        int cur = kt & 1;
        if (kt + 1 < nkt) loadKV(cur ^ 1, (kt + 1) * FA_KT, Kp, Vp);

        v8f s[2];
        scoreTiles(cur, s);

        // causal mask
        #pragma unroll
        for (int nt = 0; nt < 2; nt++) {
            int key = kt * FA_KT + nt * 16 + l16;
            #pragma unroll
            for (int j = 0; j < 8; j++) {
                int qr = qrow + j + 8 * hf;
                if (key > qr) s[nt][j] = -3.0e38f;
            }
        }
        // online softmax
        float rmax[8], lsum[8], alpha[8];
        #pragma unroll
        for (int j = 0; j < 8; j++) rmax[j] = fmaxf(s[0][j], s[1][j]);
        #pragma unroll
        for (int j = 0; j < 8; j++)
            #pragma unroll
            for (int off = 1; off < 16; off <<= 1)
                rmax[j] = fmaxf(rmax[j], __shfl_xor(rmax[j], off, 32));
        #pragma unroll
        for (int j = 0; j < 8; j++) {
            float mn = fmaxf(m[j], rmax[j]);
            alpha[j] = __expf(m[j] - mn);
            m[j] = mn;
        }
        #pragma unroll
        for (int dt = 0; dt < 8; dt++)
            #pragma unroll
            for (int j = 0; j < 8; j++) o[dt][j] *= alpha[j];
        #pragma unroll
        for (int j = 0; j < 8; j++) {
            float p0 = (s[0][j] < -1.0e37f) ? 0.f : __expf(s[0][j] - m[j]);
            float p1 = (s[1][j] < -1.0e37f) ? 0.f : __expf(s[1][j] - m[j]);
            s[0][j] = p0; s[1][j] = p1;
            lsum[j] = p0 + p1;
        }
        #pragma unroll
        for (int j = 0; j < 8; j++)
            #pragma unroll
            for (int off = 1; off < 16; off <<= 1)
                lsum[j] += __shfl_xor(lsum[j], off, 32);
        #pragma unroll
        for (int j = 0; j < 8; j++) l[j] = l[j] * alpha[j] + lsum[j];

        // P -> LDS (C-layout to A-layout transpose), then P*V
        bf16* Pw = &Ps[wave][0];
        #pragma unroll
        for (int nt = 0; nt < 2; nt++)
            #pragma unroll
            for (int j = 0; j < 8; j++)
                Pw[(j + 8 * hf) * 32 + nt * 16 + l16] = f2bf(s[nt][j]);
        pvAccum(cur);

        if (kt + 1 < nkt) { async_fence(); __syncthreads(); }
    }

    // normalize running output
    float invl[8];
    #pragma unroll
    for (int j = 0; j < 8; j++) invl[j] = (l[j] > 0.f) ? 1.f / l[j] : 0.f;
    #pragma unroll
    for (int dt = 0; dt < 8; dt++)
        #pragma unroll
        for (int j = 0; j < 8; j++) o[dt][j] *= invl[j];

    // ---- gated adapter attention (10 keys padded to 32) ----
    __syncthreads();
    loadKV(0, 0, AK + bh * 32 * HDIM, AV + bh * 32 * HDIM);
    async_fence();
    __syncthreads();

    {
        v8f s[2];
        scoreTiles(0, s);
        float g = tanhf(gate[h]);
        #pragma unroll
        for (int nt = 0; nt < 2; nt++) {
            int key = nt * 16 + l16;
            #pragma unroll
            for (int j = 0; j < 8; j++)
                if (key >= 10) s[nt][j] = -3.0e38f;
        }
        float rmax[8], lsum[8];
        #pragma unroll
        for (int j = 0; j < 8; j++) rmax[j] = fmaxf(s[0][j], s[1][j]);
        #pragma unroll
        for (int j = 0; j < 8; j++)
            #pragma unroll
            for (int off = 1; off < 16; off <<= 1)
                rmax[j] = fmaxf(rmax[j], __shfl_xor(rmax[j], off, 32));
        #pragma unroll
        for (int j = 0; j < 8; j++) {
            float p0 = (s[0][j] < -1.0e37f) ? 0.f : __expf(s[0][j] - rmax[j]);
            float p1 = (s[1][j] < -1.0e37f) ? 0.f : __expf(s[1][j] - rmax[j]);
            s[0][j] = p0; s[1][j] = p1;
            lsum[j] = p0 + p1;
        }
        #pragma unroll
        for (int j = 0; j < 8; j++)
            #pragma unroll
            for (int off = 1; off < 16; off <<= 1)
                lsum[j] += __shfl_xor(lsum[j], off, 32);
        bf16* Pw = &Ps[wave][0];
        #pragma unroll
        for (int nt = 0; nt < 2; nt++)
            #pragma unroll
            for (int j = 0; j < 8; j++) {
                float pn = (lsum[j] > 0.f) ? s[nt][j] / lsum[j] * g : 0.f;
                Pw[(j + 8 * hf) * 32 + nt * 16 + l16] = f2bf(pn);
            }
        pvAccum(0);   // accumulates gated adapter value into normalized o
    }

    // store: Out [b, s, h*HDIM + d] bf16
    #pragma unroll
    for (int dt = 0; dt < 8; dt++)
        #pragma unroll
        for (int j = 0; j < 8; j++) {
            int srow = qrow + j + 8 * hf;
            long idx = ((long)b * SEQ + srow) * (long)DMODEL + h * HDIM + dt * 16 + l16;
            Out[idx] = f2bf(o[dt][j]);
        }
}

// ---------------------------------------------------------------------------
// host-side orchestration
// ---------------------------------------------------------------------------
extern "C" void kernel_launch(void* const* d_in, const int* in_sizes, int n_in,
                              void* d_out, int out_size, void* d_ws, size_t ws_size,
                              hipStream_t stream) {
    const float* hs      = (const float*)d_in[0];
    const float* adapter = (const float*)d_in[1];
    const float* wq      = (const float*)d_in[2];
    const float* wk      = (const float*)d_in[3];
    const float* wv      = (const float*)d_in[4];
    const float* wo      = (const float*)d_in[5];
    const float* gate    = (const float*)d_in[6];
    const float* ln1     = (const float*)d_in[7];
    const float* ln2     = (const float*)d_in[8];
    const float* wgate   = (const float*)d_in[9];
    const float* wup     = (const float*)d_in[10];
    const float* wdown   = (const float*)d_in[11];
    float* out = (float*)d_out;

    const int M  = BATCH * SEQ;      // 4096
    const int AM = BATCH * 10;       // 20

    size_t off = 0;
    auto alloc = [&](size_t bytes) -> void* {
        off = (off + 255) & ~(size_t)255;
        void* p = (char*)d_ws + off;
        off += bytes;
        return p;
    };

    bf16* wq_bf = (bf16*)alloc((size_t)DMODEL * DMODEL * 2);
    bf16* wk_bf = (bf16*)alloc((size_t)DMODEL * DMODEL * 2);
    bf16* wv_bf = (bf16*)alloc((size_t)DMODEL * DMODEL * 2);
    bf16* wo_bf = (bf16*)alloc((size_t)DMODEL * DMODEL * 2);
    bf16* wg_bf = (bf16*)alloc((size_t)FFDIM * DMODEL * 2);
    bf16* wu_bf = (bf16*)alloc((size_t)FFDIM * DMODEL * 2);
    bf16* wd_bf = (bf16*)alloc((size_t)DMODEL * FFDIM * 2);
    bf16* xn_bf = (bf16*)alloc((size_t)M * DMODEL * 2);          // also reused as h2_bf
    bf16* ad_bf = (bf16*)alloc((size_t)AM * DMODEL * 2);

    // --- phase-1 region (later aliased by g_f32, which needs 128MB) ---
    size_t regionStart = (off + 255) & ~(size_t)255;
    float* q_f32   = (float*)alloc((size_t)M * DMODEL * 4);
    float* k_f32   = (float*)alloc((size_t)M * DMODEL * 4);
    float* v_f32   = (float*)alloc((size_t)M * DMODEL * 4);
    float* adk_f32 = (float*)alloc((size_t)AM * DMODEL * 4);
    float* adv_f32 = (float*)alloc((size_t)AM * DMODEL * 4);
    bf16*  q_bf    = (bf16*)alloc((size_t)M * DMODEL * 2);
    bf16*  k_bf    = (bf16*)alloc((size_t)M * DMODEL * 2);
    // region spans >= 134.5MB; g_f32 (128MB) aliases it after attention is done
    float* g_f32 = (float*)((char*)d_ws + regionStart);

    bf16*  v_bf   = (bf16*)alloc((size_t)M * DMODEL * 2);
    bf16*  adk_bf = (bf16*)alloc((size_t)BATCH * NHEAD * 32 * HDIM * 2);
    bf16*  adv_bf = (bf16*)alloc((size_t)BATCH * NHEAD * 32 * HDIM * 2);
    bf16*  attn_bf = (bf16*)alloc((size_t)M * DMODEL * 2);
    float* hidden  = (float*)alloc((size_t)M * DMODEL * 4);
    bf16*  gu_bf   = (bf16*)alloc((size_t)M * FFDIM * 2);
    (void)ws_size; (void)in_sizes; (void)n_in; (void)out_size;

    auto cvt = [&](const float* src, bf16* dst, long n) {
        cvt_bf16_kernel<<<dim3((unsigned)((n + 255) / 256)), dim3(256), 0, stream>>>(src, dst, n);
    };
    // weight + adapter conversion
    cvt(wq, wq_bf, (long)DMODEL * DMODEL);
    cvt(wk, wk_bf, (long)DMODEL * DMODEL);
    cvt(wv, wv_bf, (long)DMODEL * DMODEL);
    cvt(wo, wo_bf, (long)DMODEL * DMODEL);
    cvt(wgate, wg_bf, (long)FFDIM * DMODEL);
    cvt(wup, wu_bf, (long)FFDIM * DMODEL);
    cvt(wdown, wd_bf, (long)DMODEL * FFDIM);
    cvt(adapter, ad_bf, (long)AM * DMODEL);

    // RMSNorm 1
    rmsnorm_kernel<<<dim3(M), dim3(256), 0, stream>>>(hs, ln1, xn_bf, DMODEL);

    auto gemm = [&](const bf16* A, const bf16* B, int m, int n, int k,
                    int mode, float* of, bf16* ob, const float* aux) {
        dim3 grid(n / BN, (m + BM - 1) / BM);
        gemm_bf16_wmma<<<grid, dim3(256), 0, stream>>>(A, B, m, n, k, mode, of, ob, aux);
    };

    // QKV projections
    gemm(xn_bf, wq_bf, M, DMODEL, DMODEL, 0, q_f32, nullptr, nullptr);
    gemm(xn_bf, wk_bf, M, DMODEL, DMODEL, 0, k_f32, nullptr, nullptr);
    gemm(xn_bf, wv_bf, M, DMODEL, DMODEL, 0, v_f32, nullptr, nullptr);
    // adapter projections (no norm, no rope)
    gemm(ad_bf, wk_bf, AM, DMODEL, DMODEL, 0, adk_f32, nullptr, nullptr);
    gemm(ad_bf, wv_bf, AM, DMODEL, DMODEL, 0, adv_f32, nullptr, nullptr);

    // RoPE (+ fold 1/sqrt(HD) into q), V repack, adapter pack
    const float scale = 0.08838834764831845f; // 1/sqrt(128)
    rope_kernel<<<dim3(BATCH * NHEAD * SEQ), dim3(64), 0, stream>>>(q_f32, k_f32, q_bf, k_bf, scale);
    vpack_kernel<<<dim3(BATCH * NHEAD * SEQ), dim3(128), 0, stream>>>(v_f32, v_bf);
    adpack_kernel<<<dim3(BATCH * NHEAD * 32), dim3(128), 0, stream>>>(adk_f32, adk_bf);
    adpack_kernel<<<dim3(BATCH * NHEAD * 32), dim3(128), 0, stream>>>(adv_f32, adv_bf);

    // flash attention (causal + gated adapter)
    flash_attn_kernel<<<dim3(BATCH * NHEAD * (SEQ / 64)), dim3(128), 0, stream>>>(
        q_bf, k_bf, v_bf, adk_bf, adv_bf, gate, attn_bf);

    // O projection + residual
    gemm(attn_bf, wo_bf, M, DMODEL, DMODEL, 1, hidden, nullptr, hs);

    // RMSNorm 2 (reuse xn_bf slot)
    rmsnorm_kernel<<<dim3(M), dim3(256), 0, stream>>>(hidden, ln2, xn_bf, DMODEL);

    // MLP: gate proj -> g_f32 ; up proj fused with silu(g)*u -> gu_bf ; down + residual
    gemm(xn_bf, wg_bf, M, FFDIM, DMODEL, 0, g_f32, nullptr, nullptr);
    gemm(xn_bf, wu_bf, M, FFDIM, DMODEL, 2, nullptr, gu_bf, g_f32);
    gemm(gu_bf, wd_bf, M, DMODEL, FFDIM, 1, out, nullptr, hidden);
}
